// enc_mtan_rnn_64063732187232
// MI455X (gfx1250) — matrix operand order repaired and checked by reading, not compiled, not run on hardware
//
#include <hip/hip_runtime.h>
#include <hip/hip_bf16.h>
#include <math.h>

// ---------------------------------------------------------------------------
// MI455X (gfx1250) implementation of the mTAN encoder reference.
// Dense layers run on v_wmma_f32_16x16x32_f16 (f16 operands, fp32 accumulate).
// All GEMM operands are arranged k-contiguous so each fragment is two 16-byte
// vector loads; the V matrix is produced transposed by the previous GEMM's
// epilogue so the P@V product also reads contiguously.
// ---------------------------------------------------------------------------

typedef __attribute__((ext_vector_type(16))) _Float16 v16h;
typedef __attribute__((ext_vector_type(8)))  _Float16 v8h;
typedef __attribute__((ext_vector_type(8)))  float    v8f;

#define WPB 8  // waves per block (256 threads, wave32)

struct GemmP {
  const _Float16* A;   // (M,K) row-major, row stride lda
  const _Float16* Bm;  // element (k,n) at k + n*sbn  (k-contiguous!)
  const float*    bias;
  const float*    C0;  // optional residual, row stride ldc0
  float*          C;   // optional f32 out, row stride ldc
  _Float16*       Ch;  // optional f16 out, row stride ldch
  _Float16*       ChT; // optional f16 transposed out: [n*ldcht + m]
  int M, N, K;
  int lda, sbn, ldc, ldc0, ldch, ldcht;
  long aB, aH, bB, bH, cB, cH, c0B, c0H, chB, chH, chtB, chtH;
  int H;        // inner batch count (heads); z -> (z/H, z%H)
  float alpha;
  int act;      // 0 none, 1 relu, 2 sigmoid   out = C0 + act(alpha*acc + bias)
};

__global__ __launch_bounds__(256) void kGemm(GemmP p) {
  const int z  = blockIdx.z;
  const int bo = z / p.H, hi = z % p.H;
  const _Float16* A  = p.A  + bo * p.aB + hi * p.aH;
  const _Float16* Bm = p.Bm + bo * p.bB + hi * p.bH;
  const int lane = threadIdx.x & 31;
  const int wave = threadIdx.x >> 5;
  const int nt    = p.N >> 4;
  const int tiles = (p.M >> 4) * nt;
  const int tile  = blockIdx.x * WPB + wave;
  if (tile >= tiles) return;
  const int m0   = (tile / nt) << 4;
  const int n0   = (tile % nt) << 4;
  const int half = lane >> 4;   // K-half select (A) / K-group (B)
  const int l16  = lane & 15;
  const int am   = m0 + l16;    // A: lanes 0-15 & 16-31 both carry M=0..15
  const int bn   = n0 + l16;    // B/C: column = lane % 16

  const _Float16* Abase = A  + (long)am * p.lda;
  const _Float16* Bbase = Bm + (long)bn * p.sbn;

  // One 16x16x32 WMMA step. A 16x32 fragment: lane half selects K-halves
  // {8h..8h+7, 16+8h..16+8h+7}; B 32x16 fragment: same K-halves as A.
  auto step = [&](int kk, v8f acc) -> v8f {
    const v8h a0 = *(const v8h*)(Abase + kk + half * 8);
    const v8h a1 = *(const v8h*)(Abase + kk + 16 + half * 8);
    const v8h b0 = *(const v8h*)(Bbase + kk + half * 8);
    const v8h b1 = *(const v8h*)(Bbase + kk + 16 + half * 8);
    v16h a = __builtin_shufflevector(a0, a1, 0, 1, 2, 3, 4, 5, 6, 7,
                                     8, 9, 10, 11, 12, 13, 14, 15);
    v16h b = __builtin_shufflevector(b0, b1, 0, 1, 2, 3, 4, 5, 6, 7,
                                     8, 9, 10, 11, 12, 13, 14, 15);
    return __builtin_amdgcn_wmma_f32_16x16x32_f16(false, a, false, b, (short)0,
                                                  acc, false, false);
  };

  v8f acc0 = {}, acc1 = {};
  int kk = 0;
  for (; kk + 64 <= p.K; kk += 64) {   // 2 independent accumulators
    __builtin_prefetch(Abase + kk + 64, 0, 1);  // speculative: OOB dropped
    __builtin_prefetch(Bbase + kk + 64, 0, 1);
    acc0 = step(kk, acc0);
    acc1 = step(kk + 32, acc1);
  }
  for (; kk < p.K; kk += 32) acc0 = step(kk, acc0);
  v8f acc = acc0 + acc1;

  const float*    C0  = p.C0  ? p.C0  + bo * p.c0B  + hi * p.c0H  : nullptr;
  float*          C   = p.C   ? p.C   + bo * p.cB   + hi * p.cH   : nullptr;
  _Float16*       Ch  = p.Ch  ? p.Ch  + bo * p.chB  + hi * p.chH  : nullptr;
  _Float16*       ChT = p.ChT ? p.ChT + bo * p.chtB + hi * p.chtH : nullptr;
  const int n = n0 + l16;
  const float bv = p.bias ? p.bias[n] : 0.f;
#pragma unroll
  for (int r = 0; r < 8; ++r) {
    const int m = m0 + half * 8 + r;   // C/D: row = 8*(lane/16) + vgpr
    float v = p.alpha * acc[r] + bv;
    if (p.act == 1)      v = v > 0.f ? v : 0.f;
    else if (p.act == 2) v = 1.f / (1.f + __expf(-v));
    if (C0)  v += C0[(long)m * p.ldc0 + n];
    if (C)   C  [(long)m * p.ldc   + n] = v;
    if (Ch)  Ch [(long)m * p.ldch  + n] = (_Float16)v;
    if (ChT) ChT[(long)n * p.ldcht + m] = (_Float16)v;
  }
}

// Row-wise softmax (f32 in -> f16 probs out), one wave per row.
__global__ __launch_bounds__(256) void kSoftmax(const float* s, _Float16* pout,
                                                int rows, int n) {
  const int row  = blockIdx.x * (blockDim.x >> 5) + (threadIdx.x >> 5);
  const int lane = threadIdx.x & 31;
  if (row >= rows) return;
  const float* sr = s + (long)row * n;
  float mx = -1e30f;
  for (int i = lane; i < n; i += 32) mx = fmaxf(mx, sr[i]);
  for (int o = 16; o > 0; o >>= 1) mx = fmaxf(mx, __shfl_xor(mx, o, 32));
  float sum = 0.f;
  for (int i = lane; i < n; i += 32) sum += __expf(sr[i] - mx);
  for (int o = 16; o > 0; o >>= 1) sum += __shfl_xor(sum, o, 32);
  const float inv = 1.f / sum;
  _Float16* pr = pout + (long)row * n;
  for (int i = lane; i < n; i += 32) pr[i] = (_Float16)(__expf(sr[i] - mx) * inv);
}

// fixed_time_embedding: interleaved sin/cos, E=128, scale 48, base 10.
__global__ void kTimeEmb(const float* pos, int stride, int rows, _Float16* out) {
  const int idx = blockIdx.x * blockDim.x + threadIdx.x;
  if (idx >= rows * 64) return;
  const int row = idx >> 6, i = idx & 63;
  const float div = __expf((float)(2 * i) * (-2.302585093f / 128.f));
  const float ang = 48.f * pos[(long)row * stride] * div;
  out[(long)row * 128 + 2 * i]     = (_Float16)__sinf(ang);
  out[(long)row * 128 + 2 * i + 1] = (_Float16)__cosf(ang);
}

// masked per-channel mTAN softmax + apply: thread per (b,q,c)
// mask[b,k,c] = outp[b,k,32+(c&31)],  v[b,k,c] = outp[b,k,c],  ld(outp)=80.
__global__ void kMtan(const float* s, const float* outp, _Float16* att) {
  const int idx = blockIdx.x * blockDim.x + threadIdx.x;
  if (idx >= 32 * 128 * 64) return;
  const int c = idx & 63, q = (idx >> 6) & 127, b = idx >> 13;
  const float* sr = s + ((long)b * 128 + q) * 128;
  const float* op = outp + (long)b * 128 * 80;
  const int mc = 32 + (c & 31);
  float mx = -1e30f;
  for (int k = 0; k < 128; ++k) {
    const float sc = (op[k * 80 + mc] != 0.f) ? sr[k] : -1e9f;
    mx = fmaxf(mx, sc);
  }
  float den = 0.f, num = 0.f;
  for (int k = 0; k < 128; ++k) {
    const float sc = (op[k * 80 + mc] != 0.f) ? sr[k] : -1e9f;
    const float e  = __expf(sc - mx);
    den += e;
    num += e * op[k * 80 + c];
  }
  att[idx] = (_Float16)(num / den);
}

// Bidirectional GRU scan: block 0 = forward, block 1 = backward.
// xp = precomputed x@Wih^T + bih laid out (b*128 + t, 192). State in LDS.
__global__ __launch_bounds__(256) void kGru(const float* xpF, const float* xpB,
                                            const float* WhhF, const float* bhhF,
                                            const float* WhhB, const float* bhhB,
                                            _Float16* g16) {
  __shared__ float h[32 * 64];
  __shared__ float hp[32 * 192];
  const int dir = blockIdx.x;
  const float* xp  = dir ? xpB  : xpF;
  const float* Whh = dir ? WhhB : WhhF;
  const float* bhh = dir ? bhhB : bhhF;
  const int tid = threadIdx.x;
  for (int i = tid; i < 32 * 64; i += 256) h[i] = 0.f;
  __syncthreads();
  for (int st = 0; st < 128; ++st) {
    const int t = dir ? (127 - st) : st;
    for (int i = tid; i < 32 * 192; i += 256) {
      const int b = i / 192, g = i % 192;
      float acc = bhh[g];
      const float* wr = Whh + g * 64;
      const float* hb = h + b * 64;
      for (int j = 0; j < 64; ++j) acc += hb[j] * wr[j];
      hp[i] = acc;
    }
    __syncthreads();
    for (int i = tid; i < 32 * 64; i += 256) {
      const int b = i >> 6, j = i & 63;
      const float* xr = xp + ((long)(b * 128 + t)) * 192;
      const float r  = 1.f / (1.f + __expf(-(xr[j]       + hp[b * 192 + j])));
      const float zz = 1.f / (1.f + __expf(-(xr[64 + j]  + hp[b * 192 + 64 + j])));
      const float nn = tanhf(xr[128 + j] + r * hp[b * 192 + 128 + j]);
      const float hn = (1.f - zz) * nn + zz * h[i];
      h[i] = hn;
      g16[((long)(b * 128 + t)) * 128 + dir * 64 + j] = (_Float16)hn;
    }
    __syncthreads();
  }
}

// Pack fp32 (rows,cols,ldSrc) -> zero-padded f16 (rowsPad,ldDst).
__global__ void kPackF16(const float* src, _Float16* dst, int rows, int cols,
                         int ldSrc, int rowsPad, int ldDst) {
  const long idx = (long)blockIdx.x * blockDim.x + threadIdx.x;
  if (idx >= (long)rowsPad * ldDst) return;
  const int r = (int)(idx / ldDst), c = (int)(idx % ldDst);
  dst[idx] = (r < rows && c < cols) ? (_Float16)src[(long)r * ldSrc + c] : (_Float16)0.f;
}

__global__ void kPackBias(const float* src, float* dst, int n, int np) {
  const int i = blockIdx.x * blockDim.x + threadIdx.x;
  if (i >= np) return;
  dst[i] = (i < n) ? src[i] : 0.f;
}

// tx = concat(x, t[...,None]) -> f16 (8192, 96) zero padded.
__global__ void kBuildTx(const float* x, const float* t, _Float16* dst) {
  const long idx = (long)blockIdx.x * blockDim.x + threadIdx.x;
  if (idx >= 8192L * 96) return;
  const int r = (int)(idx / 96), c = (int)(idx % 96);
  _Float16 v = (_Float16)0.f;
  if (c < 64)       v = (_Float16)x[(long)r * 64 + c];
  else if (c == 64) v = (_Float16)t[r];
  dst[idx] = v;
}

// ---------------------------------------------------------------------------

struct Lin { _Float16* Wp; float* bp; int Kp; };
struct MabW { Lin q, k, v, o; };

extern "C" void kernel_launch(void* const* d_in, const int* in_sizes, int n_in,
                              void* d_out, int out_size, void* d_ws, size_t ws_size,
                              hipStream_t stream) {
  if (n_in < 86) return;
  auto in = [&](int i) { return (const float*)d_in[i]; };

  size_t off = 0;
  auto alloc = [&](size_t bytes) -> void* {
    void* p = (char*)d_ws + off;
    off = (off + bytes + 255) & ~size_t(255);
    return p;
  };
  auto aHp = [&](size_t n) { return (_Float16*)alloc(n * sizeof(_Float16)); };
  auto aFp = [&](size_t n) { return (float*)alloc(n * sizeof(float)); };

  auto packLin = [&](int wi, int bi, int N, int K, int Np, int Kp) -> Lin {
    Lin L; L.Kp = Kp;
    L.Wp = aHp((size_t)Np * Kp);
    L.bp = aFp((size_t)Np);
    long tot = (long)Np * Kp;
    kPackF16<<<(int)((tot + 255) / 256), 256, 0, stream>>>(in(wi), L.Wp, N, K, K, Np, Kp);
    kPackBias<<<(Np + 255) / 256, 256, 0, stream>>>(in(bi), L.bp, N, Np);
    return L;
  };
  auto packMab = [&](int base, int dq, int dk) -> MabW {
    MabW m;
    m.q = packLin(base + 0, base + 1, 128, dq, 128, dq);
    m.k = packLin(base + 2, base + 3, 128, dk, 128, dk);
    m.v = packLin(base + 4, base + 5, 128, dk, 128, dk);
    m.o = packLin(base + 6, base + 7, 128, 128, 128, 128);
    return m;
  };

  auto gemmL = [&](const _Float16* A, int lda, long aB, long aHs,
                   const _Float16* Bm, int sbn, long bB, long bHs,
                   const float* bias,
                   const float* C0, int ldc0, long c0B, long c0H,
                   float* C, int ldc, long cB, long cH,
                   _Float16* Ch, int ldch, long chB, long chH,
                   _Float16* ChT, int ldcht, long chtB, long chtH,
                   int M, int N, int K, int batch, int H, float alpha, int act) {
    GemmP p;
    p.A = A; p.Bm = Bm; p.bias = bias; p.C0 = C0; p.C = C; p.Ch = Ch; p.ChT = ChT;
    p.M = M; p.N = N; p.K = K;
    p.lda = lda; p.sbn = sbn; p.ldc = ldc; p.ldc0 = ldc0; p.ldch = ldch; p.ldcht = ldcht;
    p.aB = aB; p.aH = aHs; p.bB = bB; p.bH = bHs;
    p.cB = cB; p.cH = cH; p.c0B = c0B; p.c0H = c0H;
    p.chB = chB; p.chH = chH; p.chtB = chtB; p.chtH = chtH;
    p.H = H; p.alpha = alpha; p.act = act;
    const int tiles = (M / 16) * (N / 16);
    dim3 grid((tiles + WPB - 1) / WPB, 1, batch);
    kGemm<<<grid, 256, 0, stream>>>(p);
  };

  const float inv_sqrt_dv = 0.08838834764831845f;  // 1/sqrt(128)
  const int NB = 32;

  // ---- weight packing ----------------------------------------------------
  Lin augInit = packLin(3, 4, 256, 65, 256, 96);
  Lin augFin  = packLin(5, 6, 65, 256, 80, 256);
  MabW m00 = packMab(8, 128, 256);   // isab0.mab0
  MabW m01 = packMab(16, 256, 128);  // isab0.mab1
  MabW m10 = packMab(25, 128, 128);  // isab1.mab0
  MabW m11 = packMab(33, 128, 128);  // isab1.mab1
  MabW mpm = packMab(42, 128, 128);  // pma.mab
  MabW ms1 = packMab(50, 128, 128);  // sab1
  MabW ms2 = packMab(58, 128, 128);  // sab2
  Lin stOut = packLin(66, 67, 256, 128, 256, 128);
  Lin lq = packLin(68, 69, 128, 128, 128, 128);
  Lin lk = packLin(70, 71, 128, 128, 128, 128);
  Lin lo = packLin(72, 73, 64, 64, 64, 64);
  Lin wihF = packLin(74, 76, 192, 64, 192, 64);
  Lin wihB = packLin(78, 80, 192, 64, 192, 64);
  Lin hl1 = packLin(82, 83, 50, 128, 64, 128);
  Lin hl2 = packLin(84, 85, 32, 50, 32, 64);

  // inducing points / seeds -> f16
  _Float16* I0h = aHp(32 * 128);
  _Float16* I1h = aHp(32 * 128);
  _Float16* Sh  = aHp(128 * 128);
  kPackF16<<<(32 * 128 + 255) / 256, 256, 0, stream>>>(in(7), I0h, 32, 128, 128, 32, 128);
  kPackF16<<<(32 * 128 + 255) / 256, 256, 0, stream>>>(in(24), I1h, 32, 128, 128, 32, 128);
  kPackF16<<<(128 * 128 + 255) / 256, 256, 0, stream>>>(in(41), Sh, 128, 128, 128, 128, 128);

  // ---- persistent activations -------------------------------------------
  _Float16* tx16 = aHp(8192L * 96);
  _Float16* h16  = aHp(8192L * 256);
  _Float16* Hh16  = aHp(32L * 32 * 128);
  _Float16* X1_16 = aHp(8192L * 128);
  _Float16* Hh2_16 = aHp(32L * 32 * 128);
  _Float16* X2_16 = aHp(8192L * 128);
  _Float16* X3_16 = aHp(4096L * 128);
  _Float16* X4_16 = aHp(4096L * 128);
  _Float16* X5_16 = aHp(4096L * 128);
  _Float16* z16   = aHp(4096L * 256);
  float*    outp32 = aFp(4096L * 80);
  _Float16* kemb16 = aHp(4096L * 128);
  _Float16* qemb16 = aHp(128L * 128);
  _Float16* Q16    = aHp(128L * 128);
  _Float16* Kk16   = aHp(4096L * 128);
  float*    s32    = aFp(32L * 128 * 128);
  _Float16* attv16 = aHp(4096L * 64);
  _Float16* att2_16 = aHp(4096L * 64);
  float*    xpF32 = aFp(4096L * 192);
  float*    xpB32 = aFp(4096L * 192);
  _Float16* g16   = aHp(4096L * 128);
  _Float16* y1_16 = aHp(4096L * 64);

  // ---- mab scratch (worst-case sizes, reused) ---------------------------
  float*    Qp32  = aFp(8192L * 128);
  _Float16* Qp16  = aHp(8192L * 128);
  _Float16* Kp16  = aHp(8192L * 128);
  _Float16* VpT16 = aHp(128L * 8192);       // transposed: [feature, b*nk + k]
  float*    scr32 = aFp(128L * 32768);      // B*H * max(nq*nk)
  _Float16* prb16 = aHp(128L * 32768);
  float*    O32   = aFp(8192L * 128);
  _Float16* O16   = aHp(8192L * 128);
  (void)ws_size; (void)in_sizes; (void)out_size;

  // mab(Q, Kin): heads=4, dv=128, dh=32.  out = O + relu(O @ Wo^T + bo),
  // O = Qp + softmax(Qp Kp^T / sqrt(128)) Vp   (per (b,head))
  auto runMab = [&](const _Float16* Qf, int ldq, bool qShared, int nq,
                    const _Float16* Kf, int ldk, int nk,
                    const MabW& w, _Float16* outF) {
    const int MQ = qShared ? nq : NB * nq;
    const int ldT = NB * nk;  // transposed V leading dim
    // Qp (f32 + f16)
    gemmL(Qf, ldq, 0, 0, w.q.Wp, w.q.Kp, 0, 0, w.q.bp,
          nullptr, 0, 0, 0, Qp32, 128, 0, 0, Qp16, 128, 0, 0,
          nullptr, 0, 0, 0, MQ, 128, w.q.Kp, 1, 1, 1.f, 0);
    // Kp (f16), Vp (f16 transposed)
    gemmL(Kf, ldk, 0, 0, w.k.Wp, w.k.Kp, 0, 0, w.k.bp,
          nullptr, 0, 0, 0, nullptr, 0, 0, 0, Kp16, 128, 0, 0,
          nullptr, 0, 0, 0, NB * nk, 128, w.k.Kp, 1, 1, 1.f, 0);
    gemmL(Kf, ldk, 0, 0, w.v.Wp, w.v.Kp, 0, 0, w.v.bp,
          nullptr, 0, 0, 0, nullptr, 0, 0, 0, nullptr, 0, 0, 0,
          VpT16, ldT, 0, 0, NB * nk, 128, w.v.Kp, 1, 1, 1.f, 0);
    // scores[b,h] = Qp[b,:,h*32:] Kp[b,:,h*32:]^T / sqrt(128)
    gemmL(Qp16, 128, qShared ? 0 : (long)nq * 128, 32,
          Kp16, 128, (long)nk * 128, 32,
          nullptr, nullptr, 0, 0, 0,
          scr32, nk, (long)4 * nq * nk, (long)nq * nk,
          nullptr, 0, 0, 0, nullptr, 0, 0, 0,
          nq, nk, 32, NB * 4, 4, inv_sqrt_dv, 0);
    const int rows = NB * 4 * nq;
    kSoftmax<<<(rows + 7) / 8, 256, 0, stream>>>(scr32, prb16, rows, nk);
    // O = Qp + P V     (B operand = VpT, element (k,n) at (h*32+n)*ldT + b*nk + k)
    gemmL(prb16, nk, (long)4 * nq * nk, (long)nq * nk,
          VpT16, ldT, (long)nk, (long)32 * ldT,
          nullptr,
          Qp32, 128, qShared ? 0 : (long)nq * 128, 32,
          O32, 128, (long)nq * 128, 32,
          O16, 128, (long)nq * 128, 32,
          nullptr, 0, 0, 0,
          nq, 32, nk, NB * 4, 4, 1.f, 0);
    // out = O + relu(O Wo^T + bo)
    gemmL(O16, 128, 0, 0, w.o.Wp, 128, 0, 0, w.o.bp,
          O32, 128, 0, 0, nullptr, 0, 0, 0, outF, 128, 0, 0,
          nullptr, 0, 0, 0, NB * nq, 128, 128, 1, 1, 1.f, 1);
  };

  // ---- forward pass ------------------------------------------------------
  kBuildTx<<<(int)((8192L * 96 + 255) / 256), 256, 0, stream>>>(in(0), in(1), tx16);
  // h = tx @ Winit^T + b  (f16 out only)
  gemmL(tx16, 96, 0, 0, augInit.Wp, 96, 0, 0, augInit.bp,
        nullptr, 0, 0, 0, nullptr, 0, 0, 0, h16, 256, 0, 0,
        nullptr, 0, 0, 0, 8192, 256, 96, 1, 1, 1.f, 0);

  runMab(I0h, 128, true, 32, h16, 256, 256, m00, Hh16);     // isab0 stage 1
  runMab(h16, 256, false, 256, Hh16, 128, 32, m01, X1_16);  // isab0 stage 2
  runMab(I1h, 128, true, 32, X1_16, 128, 256, m10, Hh2_16); // isab1 stage 1
  runMab(X1_16, 128, false, 256, Hh2_16, 128, 32, m11, X2_16);
  runMab(Sh, 128, true, 128, X2_16, 128, 256, mpm, X3_16);  // PMA
  runMab(X3_16, 128, false, 128, X3_16, 128, 128, ms1, X4_16);
  runMab(X4_16, 128, false, 128, X4_16, 128, 128, ms2, X5_16);

  // z = X5 @ Wout^T + b
  gemmL(X5_16, 128, 0, 0, stOut.Wp, 128, 0, 0, stOut.bp,
        nullptr, 0, 0, 0, nullptr, 0, 0, 0, z16, 256, 0, 0,
        nullptr, 0, 0, 0, 4096, 256, 128, 1, 1, 1.f, 0);
  // outp = sigmoid(z @ Wfin^T + b)   (f32, ld 80)
  gemmL(z16, 256, 0, 0, augFin.Wp, 256, 0, 0, augFin.bp,
        nullptr, 0, 0, 0, outp32, 80, 0, 0, nullptr, 0, 0, 0,
        nullptr, 0, 0, 0, 4096, 80, 256, 1, 1, 1.f, 2);

  // time embeddings
  kTimeEmb<<<(4096 * 64 + 255) / 256, 256, 0, stream>>>(outp32 + 64, 80, 4096, kemb16);
  kTimeEmb<<<(128 * 64 + 255) / 256, 256, 0, stream>>>(in(2), 1, 128, qemb16);
  // Q = q_emb @ Wlq^T + b ; Kk = key_emb @ Wlk^T + b
  gemmL(qemb16, 128, 0, 0, lq.Wp, 128, 0, 0, lq.bp,
        nullptr, 0, 0, 0, nullptr, 0, 0, 0, Q16, 128, 0, 0,
        nullptr, 0, 0, 0, 128, 128, 128, 1, 1, 1.f, 0);
  gemmL(kemb16, 128, 0, 0, lk.Wp, 128, 0, 0, lk.bp,
        nullptr, 0, 0, 0, nullptr, 0, 0, 0, Kk16, 128, 0, 0,
        nullptr, 0, 0, 0, 4096, 128, 128, 1, 1, 1.f, 0);
  // scores[b] = Q Kk[b]^T / sqrt(128)  (Q shared across batch)
  gemmL(Q16, 128, 0, 0, Kk16, 128, 128L * 128, 0,
        nullptr, nullptr, 0, 0, 0,
        s32, 128, 128L * 128, 0, nullptr, 0, 0, 0,
        nullptr, 0, 0, 0, 128, 128, 128, 32, 1, inv_sqrt_dv, 0);
  kMtan<<<(32 * 128 * 64 + 255) / 256, 256, 0, stream>>>(s32, outp32, attv16);
  // att = attv @ Wlo^T + b
  gemmL(attv16, 64, 0, 0, lo.Wp, 64, 0, 0, lo.bp,
        nullptr, 0, 0, 0, nullptr, 0, 0, 0, att2_16, 64, 0, 0,
        nullptr, 0, 0, 0, 4096, 64, 64, 1, 1, 1.f, 0);
  // GRU input projections
  gemmL(att2_16, 64, 0, 0, wihF.Wp, 64, 0, 0, wihF.bp,
        nullptr, 0, 0, 0, xpF32, 192, 0, 0, nullptr, 0, 0, 0,
        nullptr, 0, 0, 0, 4096, 192, 64, 1, 1, 1.f, 0);
  gemmL(att2_16, 64, 0, 0, wihB.Wp, 64, 0, 0, wihB.bp,
        nullptr, 0, 0, 0, xpB32, 192, 0, 0, nullptr, 0, 0, 0,
        nullptr, 0, 0, 0, 4096, 192, 64, 1, 1, 1.f, 0);
  kGru<<<2, 256, 0, stream>>>(xpF32, xpB32, in(75), in(77), in(79), in(81), g16);
  // head
  gemmL(g16, 128, 0, 0, hl1.Wp, 128, 0, 0, hl1.bp,
        nullptr, 0, 0, 0, nullptr, 0, 0, 0, y1_16, 64, 0, 0,
        nullptr, 0, 0, 0, 4096, 64, 128, 1, 1, 1.f, 1);
  gemmL(y1_16, 64, 0, 0, hl2.Wp, 64, 0, 0, hl2.bp,
        nullptr, 0, 0, 0, (float*)d_out, 32, 0, 0, nullptr, 0, 0, 0,
        nullptr, 0, 0, 0, 4096, 32, 64, 1, 1, 1.f, 0);
}
